// SchNETBondEmbedding_12833362280995
// MI455X (gfx1250) — matrix operand order repaired and checked by reading, not compiled, not run on hardware
//
#include <hip/hip_runtime.h>

typedef __attribute__((ext_vector_type(16))) _Float16 v16h;
typedef __attribute__((ext_vector_type(8)))  _Float16 v8h;
typedef __attribute__((ext_vector_type(8)))  float    v8f;
typedef __attribute__((ext_vector_type(4)))  float    v4f;

#define DIM     64
#define NNODES  50000
#define NEDGES  800000
#define CUTOFFF 5.0f
#define LOG2F_  0.69314718055994530942f

// -------- helpers --------

__device__ __forceinline__ float sspf(float x) {
  // softplus(x) - log(2), numerically stable
  float t = __expf(-fabsf(x));
  return fmaxf(x, 0.0f) + __logf(1.0f + t) - LOG2F_;
}

__device__ __forceinline__ v8f wmma_f16(v16h a, v16h b, v8f c) {
  return __builtin_amdgcn_wmma_f32_16x16x32_f16(false, a, false, b, (short)0, c,
                                                false, false);
}

__device__ __forceinline__ void lds_fence() {
  asm volatile("s_wait_dscnt 0" ::: "memory");
}

// Native non-returning f32 atomic add (L2-resolved, STOREcnt-tracked, no CAS loop).
__device__ __forceinline__ void atomic_add_f32(float* p, float v) {
  asm volatile("global_atomic_add_f32 %0, %1, off" : : "v"(p), "v"(v) : "memory");
}

// Stage a 64x64 row-major f32 weight matrix into LDS as column-major f16.
__device__ __forceinline__ void stage_weights(_Float16* wlds, const float* __restrict__ W) {
  for (int i = threadIdx.x; i < 64 * 64; i += 256) {
    int k = i >> 6, n = i & 63;
    wlds[n * 64 + k] = (_Float16)W[i];
  }
}

// C(16x64) += A(16x64) @ W(64x64); W column-major f16 in LDS.
// acc[nt] covers columns nt*16 .. nt*16+15 (C layout: N = lane&15, M = v + 8*(lane>=16)).
__device__ __forceinline__ void gemm_stage(const _Float16* wlds, const v16h a[2],
                                           v8f acc[4], int lane) {
  int nlo  = lane & 15;
  int ksel = (lane & 16) ? 8 : 0;  // 0 or 8: B-layout K offset within the 32-K tile (second half at +16)
#pragma unroll
  for (int nt = 0; nt < 4; ++nt) {
    v8f c = acc[nt];
#pragma unroll
    for (int kt = 0; kt < 2; ++kt) {
      const _Float16* bp = wlds + (nt * 16 + nlo) * 64 + kt * 32 + ksel;
      v8h blo = *(const v8h*)bp;
      v8h bhi = *(const v8h*)(bp + 16);
      v16h bv;
#pragma unroll
      for (int e = 0; e < 8; ++e) { bv[e] = blo[e]; bv[e + 8] = bhi[e]; }
      c = wmma_f16(a[kt], bv, c);
    }
    acc[nt] = c;
  }
}

// Load A layout (16x32 f16 per ktile) from a row-major 16x64 f16 LDS scratch.
__device__ __forceinline__ void load_a_from_lds(const _Float16* hT, v16h a[2], int lane) {
  int m    = lane & 15;
  int koff = (lane & 16) ? 8 : 0;
#pragma unroll
  for (int kt = 0; kt < 2; ++kt) {
    v8h lo = *(const v8h*)(hT + m * 64 + kt * 32 + koff);
    v8h hi = *(const v8h*)(hT + m * 64 + kt * 32 + 16 + koff);
    v16h v;
#pragma unroll
    for (int e = 0; e < 8; ++e) { v[e] = lo[e]; v[e + 8] = hi[e]; }
    a[kt] = v;
  }
}

// Store C layout accumulators to row-major 16x64 f16 LDS scratch.
__device__ __forceinline__ void store_c_to_lds(_Float16* hT, const v8f acc[4], int lane) {
  int n0 = lane & 15;
  int mb = (lane & 16) ? 8 : 0;
#pragma unroll
  for (int nt = 0; nt < 4; ++nt)
#pragma unroll
    for (int v = 0; v < 8; ++v)
      hT[(mb + v) * 64 + nt * 16 + n0] = (_Float16)acc[nt][v];
}

// Add bias (per column) and optionally apply ssp, in C layout.
__device__ __forceinline__ void bias_act(v8f acc[4], const float* __restrict__ b,
                                         int lane, bool act) {
  int n0 = lane & 15;
#pragma unroll
  for (int nt = 0; nt < 4; ++nt) {
    float bv = b[nt * 16 + n0];
#pragma unroll
    for (int v = 0; v < 8; ++v) {
      float x = acc[nt][v] + bv;
      acc[nt][v] = act ? sspf(x) : x;
    }
  }
}

__device__ __forceinline__ void zero_acc(v8f acc[4]) {
#pragma unroll
  for (int nt = 0; nt < 4; ++nt) acc[nt] = {};
}

// Load + convert a 16x64 f32 row-major global tile into A layout.
__device__ __forceinline__ void load_a_global(const float* __restrict__ base,
                                              v16h a[2], int lane) {
  int m    = lane & 15;
  int koff = (lane & 16) ? 8 : 0;
  const float* row = base + m * 64;
#pragma unroll
  for (int kt = 0; kt < 2; ++kt) {
    v4f q0 = *(const v4f*)(row + kt * 32 + koff);
    v4f q1 = *(const v4f*)(row + kt * 32 + koff + 4);
    v4f r0 = *(const v4f*)(row + kt * 32 + 16 + koff);
    v4f r1 = *(const v4f*)(row + kt * 32 + 16 + koff + 4);
    v16h v;
#pragma unroll
    for (int e = 0; e < 4; ++e) {
      v[e]      = (_Float16)q0[e];
      v[e + 4]  = (_Float16)q1[e];
      v[e + 8]  = (_Float16)r0[e];
      v[e + 12] = (_Float16)r1[e];
    }
    a[kt] = v;
  }
}

// -------- kernels --------

// xi_prev[i][d] = species_W[species[i]][d]
__global__ __launch_bounds__(256) void k_embed(const int* __restrict__ species,
                                               const float* __restrict__ sW,
                                               float* __restrict__ out) {
  int i = blockIdx.x * 256 + threadIdx.x;
  if (i < NNODES * DIM) out[i] = sW[species[i >> 6] * DIM + (i & 63)];
}

// out = in @ W + b   (no activation)
__global__ __launch_bounds__(256) void k_gemm1(const float* __restrict__ in,
                                               const float* __restrict__ W,
                                               const float* __restrict__ b,
                                               float* __restrict__ out, int ntiles) {
  __shared__ _Float16 wlds[64 * 64];
  stage_weights(wlds, W);
  __syncthreads();
  int wave = threadIdx.x >> 5, lane = threadIdx.x & 31;
  int tile = blockIdx.x * 8 + wave;
  if (tile >= ntiles) return;
  v16h a[2];
  load_a_global(in + tile * 16 * DIM, a, lane);
  v8f acc[4];
  zero_acc(acc);
  gemm_stage(wlds, a, acc, lane);
  bias_act(acc, b, lane, false);
  int n0 = lane & 15, mb = (lane & 16) ? 8 : 0;
#pragma unroll
  for (int nt = 0; nt < 4; ++nt)
#pragma unroll
    for (int v = 0; v < 8; ++v)
      out[(tile * 16 + mb + v) * DIM + nt * 16 + n0] = acc[nt][v];
}

// Fused per-edge: rbf(dist) -> 3-stage MLP -> ssp -> * xi_lin[dst] * sw -> atomic scatter-add to xi_acc[src]
__global__ __launch_bounds__(256) void k_edge(
    const float* __restrict__ dist, const float* __restrict__ sw,
    const int* __restrict__ esrc, const int* __restrict__ edst,
    const float* __restrict__ xi_lin,
    const float* __restrict__ W0, const float* __restrict__ b0,
    const float* __restrict__ W1, const float* __restrict__ b1,
    const float* __restrict__ W2, const float* __restrict__ b2,
    float* __restrict__ xi_acc) {
  __shared__ _Float16 w0[4096], w1[4096], w2[4096];
  __shared__ _Float16 hT[8][16 * 64];
  stage_weights(w0, W0);
  stage_weights(w1, W1);
  stage_weights(w2, W2);
  __syncthreads();

  int wave = threadIdx.x >> 5, lane = threadIdx.x & 31;
  int tile = blockIdx.x * 8 + wave;  // NEDGES/16 tiles exactly
  _Float16* myh = hT[wave];

  int m      = lane & 15;
  int koff   = (lane & 16) ? 8 : 0;
  float d    = dist[tile * 16 + m];
  const float step = CUTOFFF / 63.0f;                              // mu spacing
  const float coef = (63.0f * 63.0f) / (2.0f * CUTOFFF * CUTOFFF); // 1/(2 sigma^2)

  // Build RBF A-tile directly in registers (A layout).
  v16h a[2];
#pragma unroll
  for (int kt = 0; kt < 2; ++kt) {
    v16h v;
#pragma unroll
    for (int e = 0; e < 16; ++e) {
      int K    = kt * 32 + ((e & 8) ? 16 : 0) + koff + (e & 7);
      float dd = d - (float)K * step;
      v[e] = (_Float16)__expf(-dd * dd * coef);
    }
    a[kt] = v;
  }

  v8f acc[4];
  // stage 0: h = ssp(rbf @ W0 + b0)
  zero_acc(acc);
  gemm_stage(w0, a, acc, lane);
  bias_act(acc, b0, lane, true);
  store_c_to_lds(myh, acc, lane);
  lds_fence();
  load_a_from_lds(myh, a, lane);

  // stage 1: h = ssp(h @ W1 + b1)
  zero_acc(acc);
  gemm_stage(w1, a, acc, lane);
  bias_act(acc, b1, lane, true);
  store_c_to_lds(myh, acc, lane);
  lds_fence();
  load_a_from_lds(myh, a, lane);

  // stage 2: ssp(h @ W2 + b2)
  zero_acc(acc);
  gemm_stage(w2, a, acc, lane);
  bias_act(acc, b2, lane, true);

  // gather xi_lin[edge_dst], scale by switch, scatter-add to xi_acc[edge_src]
  int n0 = lane & 15, mb = (lane & 16) ? 8 : 0;
  int   srcs[8], dsts[8];
  float sws[8];
#pragma unroll
  for (int v = 0; v < 8; ++v) {
    int e   = tile * 16 + mb + v;
    srcs[v] = esrc[e];
    dsts[v] = edst[e];
    sws[v]  = sw[e];
  }
#pragma unroll
  for (int nt = 0; nt < 4; ++nt) {
    int col = nt * 16 + n0;
#pragma unroll
    for (int v = 0; v < 8; ++v) {
      float msg = acc[nt][v] * xi_lin[dsts[v] * DIM + col] * sws[v];
      atomic_add_f32(&xi_acc[srcs[v] * DIM + col], msg);
    }
  }
}

// xi_prev += (ssp(acc @ W2 + b2)) @ W3 + b3
__global__ __launch_bounds__(256) void k_gemm23(const float* __restrict__ acc_in,
                                                const float* __restrict__ W2,
                                                const float* __restrict__ b2,
                                                const float* __restrict__ W3,
                                                const float* __restrict__ b3,
                                                float* __restrict__ xi_prev, int ntiles) {
  __shared__ _Float16 wa[4096], wb[4096];
  __shared__ _Float16 hT[8][16 * 64];
  stage_weights(wa, W2);
  stage_weights(wb, W3);
  __syncthreads();
  int wave = threadIdx.x >> 5, lane = threadIdx.x & 31;
  int tile = blockIdx.x * 8 + wave;
  if (tile >= ntiles) return;
  _Float16* myh = hT[wave];

  v16h a[2];
  load_a_global(acc_in + tile * 16 * DIM, a, lane);
  v8f acc[4];
  zero_acc(acc);
  gemm_stage(wa, a, acc, lane);
  bias_act(acc, b2, lane, true);
  store_c_to_lds(myh, acc, lane);
  lds_fence();
  load_a_from_lds(myh, a, lane);

  zero_acc(acc);
  gemm_stage(wb, a, acc, lane);
  bias_act(acc, b3, lane, false);

  int n0 = lane & 15, mb = (lane & 16) ? 8 : 0;
#pragma unroll
  for (int nt = 0; nt < 4; ++nt)
#pragma unroll
    for (int v = 0; v < 8; ++v) {
      int idx = (tile * 16 + mb + v) * DIM + nt * 16 + n0;
      xi_prev[idx] += acc[nt][v];
    }
}

// -------- launch --------

extern "C" void kernel_launch(void* const* d_in, const int* in_sizes, int n_in,
                              void* d_out, int out_size, void* d_ws, size_t ws_size,
                              hipStream_t stream) {
  const int*   species = (const int*)d_in[0];
  const int*   esrc    = (const int*)d_in[1];
  const int*   edst    = (const int*)d_in[2];
  const float* dist    = (const float*)d_in[3];
  const float* sw      = (const float*)d_in[4];
  const float* sW      = (const float*)d_in[5];
  const float* aw1W    = (const float*)d_in[6];
  const float* aw1b    = (const float*)d_in[7];
  const float* aw2W    = (const float*)d_in[8];
  const float* aw2b    = (const float*)d_in[9];
  const float* aw3W    = (const float*)d_in[10];
  const float* aw3b    = (const float*)d_in[11];
  const float* fW0     = (const float*)d_in[12];
  const float* fb0     = (const float*)d_in[13];
  const float* fW1     = (const float*)d_in[14];
  const float* fb1     = (const float*)d_in[15];
  const float* fW2     = (const float*)d_in[16];
  const float* fb2     = (const float*)d_in[17];

  float* xi     = (float*)d_out;                 // xi_prev, updated in place
  float* xi_lin = (float*)d_ws;                  // 50000*64 f32
  float* xi_acc = xi_lin + (size_t)NNODES * DIM; // 50000*64 f32

  k_embed<<<(NNODES * DIM + 255) / 256, 256, 0, stream>>>(species, sW, xi);

  const int ntiles   = NNODES / 16;        // 3125
  const int ngblocks = (ntiles + 7) / 8;   // 391
  const int eblocks  = NEDGES / 16 / 8;    // 6250

  for (int l = 0; l < 3; ++l) {
    k_gemm1<<<ngblocks, 256, 0, stream>>>(xi, aw1W + l * 4096, aw1b + l * 64,
                                          xi_lin, ntiles);
    hipMemsetAsync(xi_acc, 0, (size_t)NNODES * DIM * sizeof(float), stream);
    k_edge<<<eblocks, 256, 0, stream>>>(dist, sw, esrc, edst, xi_lin,
                                        fW0 + l * 4096, fb0 + l * 64,
                                        fW1 + l * 4096, fb1 + l * 64,
                                        fW2 + l * 4096, fb2 + l * 64, xi_acc);
    k_gemm23<<<ngblocks, 256, 0, stream>>>(xi_acc, aw2W + l * 4096, aw2b + l * 64,
                                           aw3W + l * 4096, aw3b + l * 64, xi,
                                           ntiles);
  }
}
